// MoEFFN_29300266893973
// MI455X (gfx1250) — compile-verified, matrix operand order checked by reading, not run on hardware
//
#include <hip/hip_runtime.h>
#include <hip/hip_bf16.h>
#include <math.h>

typedef __attribute__((ext_vector_type(16))) __bf16 v16bf;
typedef __attribute__((ext_vector_type(8)))  __bf16 v8bf;
typedef __attribute__((ext_vector_type(8)))  float  v8f;

static constexpr int S_TOK  = 2048;
static constexpr int C_DIM  = 1024;
static constexpr int H_DIM  = 256;
static constexpr int HS_DIM = 2048;
static constexpr int E_NUM  = 16;

// ------------------------------------------------------------------
// f32 -> bf16 conversion (vectorized x4)
// ------------------------------------------------------------------
__global__ __launch_bounds__(256)
void cvt_f32_bf16(const float* __restrict__ src, __bf16* __restrict__ dst, int n) {
    int i = (blockIdx.x * 256 + threadIdx.x) * 4;
    if (i + 3 < n) {
        float4 v = *(const float4*)(src + i);
        dst[i + 0] = (__bf16)v.x;
        dst[i + 1] = (__bf16)v.y;
        dst[i + 2] = (__bf16)v.z;
        dst[i + 3] = (__bf16)v.w;
    }
}

// ------------------------------------------------------------------
// Router: sigmoid scores, grouped top-k (2 of 4 groups, top-4 experts),
// renormalized unbiased weights -> dense combine [S, E]
// One wave32 per token.
// ------------------------------------------------------------------
__global__ __launch_bounds__(32)
void router_kernel(const float* __restrict__ x, const float* __restrict__ rw,
                   const float* __restrict__ bias, float* __restrict__ combine) {
    int s = blockIdx.x;
    int t = threadIdx.x;
    __shared__ float part[32];
    __shared__ float scores[16];
    __shared__ float wout[16];

    int e = t & 15, half = t >> 4;
    const float* xr = x  + (long)s * C_DIM + half * (C_DIM / 2);
    const float* wr = rw + (long)e * C_DIM + half * (C_DIM / 2);
    float acc = 0.f;
    for (int c = 0; c < C_DIM / 2; ++c) acc += xr[c] * wr[c];
    part[t] = acc;
    __syncthreads();
    if (t < 16) {
        float logit = part[t] + part[t + 16];
        scores[t] = 1.f / (1.f + expf(-logit));
    }
    __syncthreads();
    if (t == 0) {
        float sc[16], sb[16];
        for (int i = 0; i < 16; ++i) { sc[i] = scores[i]; sb[i] = sc[i] + bias[i]; }
        // group score = sum of top-2 biased scores within each group of 4
        float gs[4];
        for (int g = 0; g < 4; ++g) {
            float m1 = -1e30f, m2 = -1e30f;
            for (int j = 0; j < 4; ++j) {
                float v = sb[g * 4 + j];
                if (v > m1) { m2 = m1; m1 = v; } else if (v > m2) m2 = v;
            }
            gs[g] = m1 + m2;
        }
        int g1 = 0;
        for (int g = 1; g < 4; ++g) if (gs[g] > gs[g1]) g1 = g;
        int g2 = -1;
        for (int g = 0; g < 4; ++g) { if (g == g1) continue; if (g2 < 0 || gs[g] > gs[g2]) g2 = g; }
        float tmp[16];
        for (int i = 0; i < 16; ++i) {
            int g = i >> 2;
            tmp[i] = (g == g1 || g == g2) ? sb[i] : -1e30f;
        }
        int idx[4];
        for (int k = 0; k < 4; ++k) {
            int best = 0;
            for (int i = 1; i < 16; ++i) if (tmp[i] > tmp[best]) best = i;
            idx[k] = best; tmp[best] = -1e30f;
        }
        float wsum = 1e-20f;
        for (int k = 0; k < 4; ++k) wsum += sc[idx[k]];
        for (int i = 0; i < 16; ++i) wout[i] = 0.f;
        for (int k = 0; k < 4; ++k) wout[idx[k]] = sc[idx[k]] / wsum;
    }
    __syncthreads();
    if (t < 16) combine[(long)s * E_NUM + t] = wout[t];
}

// ------------------------------------------------------------------
// Fused gate/up dual-GEMM with SiLU epilogue (bf16 WMMA, f32 acc).
//   H[m,n] = bf16( silu(A@Bg) * (A@Bu) * combine[m,e] )
// BT=true : B stored [N,K] row-major (ldb = K)   -- shared expert weights
// BT=false: B stored [K,N] row-major (ldb = N)   -- per-expert weights
// gridDim.z = expert batch (strideB/strideH element strides).
// Block tile 64x128, 8 waves, wave tile 32x32 (2x2 WMMA).
// ------------------------------------------------------------------
template <bool BT>
__global__ __launch_bounds__(256)
void gateup_kernel(const __bf16* __restrict__ A, int lda,
                   const __bf16* __restrict__ Bg, const __bf16* __restrict__ Bu,
                   int ldb, long strideB,
                   __bf16* __restrict__ Hout, int ldh, long strideH,
                   const float* __restrict__ combine, int K) {
    constexpr int BM = 64, BN = 128, BK = 32;
    __shared__ __bf16 As [BM][BK];
    __shared__ __bf16 Bgs[BN][BK];
    __shared__ __bf16 Bus[BN][BK];

    const int e  = blockIdx.z;
    const __bf16* bg = Bg + (long)e * strideB;
    const __bf16* bu = Bu + (long)e * strideB;
    __bf16* hout = Hout + (long)e * strideH;

    const int m0 = blockIdx.y * BM;
    const int n0 = blockIdx.x * BN;
    const int tid  = threadIdx.x;
    const int lane = tid & 31, wid = tid >> 5;
    const int wm = (wid >> 2) * 32, wn = (wid & 3) * 32;
    const int l15 = lane & 15, lhi = lane >> 4;

    v8f accG[2][2] = {{{0}}};
    v8f accU[2][2] = {{{0}}};

    for (int k0 = 0; k0 < K; k0 += BK) {
        { // A tile [64][32]: 8 bf16 per thread
            int r = tid >> 2, c = (tid & 3) * 8;
            *(uint4*)&As[r][c] = *(const uint4*)(A + (long)(m0 + r) * lda + k0 + c);
            if (k0 + BK < K)
                __builtin_prefetch(A + (long)(m0 + r) * lda + k0 + BK + c, 0, 1);
        }
        if (BT) { // B [N,K]: contiguous along K
            int n = tid >> 1, c = (tid & 1) * 16;
            *(uint4*)&Bgs[n][c]     = *(const uint4*)(bg + (long)(n0 + n) * ldb + k0 + c);
            *(uint4*)&Bgs[n][c + 8] = *(const uint4*)(bg + (long)(n0 + n) * ldb + k0 + c + 8);
            *(uint4*)&Bus[n][c]     = *(const uint4*)(bu + (long)(n0 + n) * ldb + k0 + c);
            *(uint4*)&Bus[n][c + 8] = *(const uint4*)(bu + (long)(n0 + n) * ldb + k0 + c + 8);
        } else {  // B [K,N]: transpose into N-major LDS
            int kk = tid & 31, nc = (tid >> 5) * 16;
            uint4 vg0 = *(const uint4*)(bg + (long)(k0 + kk) * ldb + n0 + nc);
            uint4 vg1 = *(const uint4*)(bg + (long)(k0 + kk) * ldb + n0 + nc + 8);
            uint4 vu0 = *(const uint4*)(bu + (long)(k0 + kk) * ldb + n0 + nc);
            uint4 vu1 = *(const uint4*)(bu + (long)(k0 + kk) * ldb + n0 + nc + 8);
            const __bf16* pg0 = (const __bf16*)&vg0;
            const __bf16* pg1 = (const __bf16*)&vg1;
            const __bf16* pu0 = (const __bf16*)&vu0;
            const __bf16* pu1 = (const __bf16*)&vu1;
            for (int j = 0; j < 8; ++j) {
                Bgs[nc + j][kk]     = pg0[j];
                Bgs[nc + 8 + j][kk] = pg1[j];
                Bus[nc + j][kk]     = pu0[j];
                Bus[nc + 8 + j][kk] = pu1[j];
            }
        }
        __syncthreads();

        // ISA A-frag layout: lanes 0-15 hold K 0-7 / 16-23, lanes 16-31 K 8-15 / 24-31
        v16bf aF[2], gF[2], uF[2];
        for (int i = 0; i < 2; ++i) {
            int row = wm + i * 16 + l15;
            ((v8bf*)&aF[i])[0] = *(const v8bf*)&As[row][lhi * 8];
            ((v8bf*)&aF[i])[1] = *(const v8bf*)&As[row][lhi * 8 + 16];
        }
        for (int j = 0; j < 2; ++j) {
            int n = wn + j * 16 + l15;
            ((v8bf*)&gF[j])[0] = *(const v8bf*)&Bgs[n][lhi * 8];
            ((v8bf*)&gF[j])[1] = *(const v8bf*)&Bgs[n][lhi * 8 + 16];
            ((v8bf*)&uF[j])[0] = *(const v8bf*)&Bus[n][lhi * 8];
            ((v8bf*)&uF[j])[1] = *(const v8bf*)&Bus[n][lhi * 8 + 16];
        }
        for (int i = 0; i < 2; ++i)
            for (int j = 0; j < 2; ++j) {
                accG[i][j] = __builtin_amdgcn_wmma_f32_16x16x32_bf16(
                    false, aF[i], false, gF[j], (short)0, accG[i][j], false, false);
                accU[i][j] = __builtin_amdgcn_wmma_f32_16x16x32_bf16(
                    false, aF[i], false, uF[j], (short)0, accU[i][j], false, false);
            }
        __syncthreads();
    }

    // Epilogue: h = silu(g) * u * combine   (C/D layout: vgpr r -> row r or r+8)
    for (int i = 0; i < 2; ++i)
        for (int j = 0; j < 2; ++j)
            for (int r = 0; r < 8; ++r) {
                int m = m0 + wm + i * 16 + lhi * 8 + r;
                int n = n0 + wn + j * 16 + l15;
                float g = accG[i][j][r];
                float u = accU[i][j][r];
                float h = (g / (1.f + expf(-g))) * u;
                if (combine) h *= combine[(long)m * E_NUM + e];
                hout[(long)m * ldh + n] = (__bf16)h;
            }
}

// ------------------------------------------------------------------
// Down-projection GEMM (bf16 WMMA, f32 out).
//   Out[m,n] (+)= sum_e  A_e[m,:K] @ B_e
// BT as above. nE experts looped INSIDE the block (tile ownership ->
// race-free accumulation, no atomics). ACC=true reads existing Out.
// ------------------------------------------------------------------
template <bool BT, bool ACC>
__global__ __launch_bounds__(256)
void down_kernel(const __bf16* __restrict__ A, int lda, long strideA,
                 const __bf16* __restrict__ B, int ldb, long strideB,
                 float* __restrict__ Out, int ldc, int K, int nE) {
    constexpr int BM = 64, BN = 128, BK = 32;
    __shared__ __bf16 As[BM][BK];
    __shared__ __bf16 Bs[BN][BK];

    const int m0 = blockIdx.y * BM;
    const int n0 = blockIdx.x * BN;
    const int tid  = threadIdx.x;
    const int lane = tid & 31, wid = tid >> 5;
    const int wm = (wid >> 2) * 32, wn = (wid & 3) * 32;
    const int l15 = lane & 15, lhi = lane >> 4;

    v8f acc[2][2] = {{{0}}};

    for (int e = 0; e < nE; ++e) {
        const __bf16* a = A + (long)e * strideA;
        const __bf16* b = B + (long)e * strideB;
        for (int k0 = 0; k0 < K; k0 += BK) {
            {
                int r = tid >> 2, c = (tid & 3) * 8;
                *(uint4*)&As[r][c] = *(const uint4*)(a + (long)(m0 + r) * lda + k0 + c);
            }
            if (BT) {
                int n = tid >> 1, c = (tid & 1) * 16;
                *(uint4*)&Bs[n][c]     = *(const uint4*)(b + (long)(n0 + n) * ldb + k0 + c);
                *(uint4*)&Bs[n][c + 8] = *(const uint4*)(b + (long)(n0 + n) * ldb + k0 + c + 8);
            } else {
                int kk = tid & 31, nc = (tid >> 5) * 16;
                uint4 v0 = *(const uint4*)(b + (long)(k0 + kk) * ldb + n0 + nc);
                uint4 v1 = *(const uint4*)(b + (long)(k0 + kk) * ldb + n0 + nc + 8);
                const __bf16* p0 = (const __bf16*)&v0;
                const __bf16* p1 = (const __bf16*)&v1;
                for (int j = 0; j < 8; ++j) {
                    Bs[nc + j][kk]     = p0[j];
                    Bs[nc + 8 + j][kk] = p1[j];
                }
            }
            __syncthreads();

            v16bf aF[2], bF[2];
            for (int i = 0; i < 2; ++i) {
                int row = wm + i * 16 + l15;
                ((v8bf*)&aF[i])[0] = *(const v8bf*)&As[row][lhi * 8];
                ((v8bf*)&aF[i])[1] = *(const v8bf*)&As[row][lhi * 8 + 16];
            }
            for (int j = 0; j < 2; ++j) {
                int n = wn + j * 16 + l15;
                ((v8bf*)&bF[j])[0] = *(const v8bf*)&Bs[n][lhi * 8];
                ((v8bf*)&bF[j])[1] = *(const v8bf*)&Bs[n][lhi * 8 + 16];
            }
            for (int i = 0; i < 2; ++i)
                for (int j = 0; j < 2; ++j)
                    acc[i][j] = __builtin_amdgcn_wmma_f32_16x16x32_bf16(
                        false, aF[i], false, bF[j], (short)0, acc[i][j], false, false);
            __syncthreads();
        }
    }

    for (int i = 0; i < 2; ++i)
        for (int j = 0; j < 2; ++j)
            for (int r = 0; r < 8; ++r) {
                int m = m0 + wm + i * 16 + lhi * 8 + r;
                int n = n0 + wn + j * 16 + l15;
                float v = acc[i][j][r];
                if (ACC) v += Out[(long)m * ldc + n];
                Out[(long)m * ldc + n] = v;
            }
}

// ------------------------------------------------------------------
// Host orchestration
// ------------------------------------------------------------------
extern "C" void kernel_launch(void* const* d_in, const int* in_sizes, int n_in,
                              void* d_out, int out_size, void* d_ws, size_t ws_size,
                              hipStream_t stream) {
    const float* x   = (const float*)d_in[0];
    const float* rw  = (const float*)d_in[1];
    const float* cb  = (const float*)d_in[2];
    const float* gw  = (const float*)d_in[3];
    const float* uw  = (const float*)d_in[4];
    const float* dw  = (const float*)d_in[5];
    const float* sgw = (const float*)d_in[6];
    const float* suw = (const float*)d_in[7];
    const float* sdw = (const float*)d_in[8];
    float* out = (float*)d_out;

    char* ws = (char*)d_ws;
    size_t off = 0;
    auto take = [&](size_t bytes) { size_t o = off; off = (off + bytes + 255) & ~(size_t)255; return o; };

    const long nX  = (long)S_TOK * C_DIM;
    const long nSW = (long)HS_DIM * C_DIM;        // shared gate/up; down is C*HS = same count
    const long nGW = (long)E_NUM * C_DIM * H_DIM; // routed gate/up
    const long nDW = (long)E_NUM * H_DIM * C_DIM; // routed down
    const long nHS = (long)S_TOK * HS_DIM;
    const long nH  = (long)E_NUM * S_TOK * H_DIM;

    float*  combine = (float*)(ws + take((size_t)S_TOK * E_NUM * 4));
    __bf16* xbf  = (__bf16*)(ws + take(nX  * 2));
    __bf16* sgwb = (__bf16*)(ws + take(nSW * 2));
    __bf16* suwb = (__bf16*)(ws + take(nSW * 2));
    __bf16* sdwb = (__bf16*)(ws + take(nSW * 2));
    __bf16* gwb  = (__bf16*)(ws + take(nGW * 2));
    __bf16* uwb  = (__bf16*)(ws + take(nGW * 2));
    __bf16* dwb  = (__bf16*)(ws + take(nDW * 2));
    __bf16* hsbf = (__bf16*)(ws + take(nHS * 2));
    __bf16* hbf  = (__bf16*)(ws + take(nH  * 2));

    // 1) router
    router_kernel<<<S_TOK, 32, 0, stream>>>(x, rw, cb, combine);

    // 2) conversions to bf16
    auto cvt = [&](const float* s, __bf16* d, long n) {
        cvt_f32_bf16<<<(int)((n / 4 + 255) / 256), 256, 0, stream>>>(s, d, (int)n);
    };
    cvt(x, xbf, nX);
    cvt(sgw, sgwb, nSW);
    cvt(suw, suwb, nSW);
    cvt(sdw, sdwb, nSW);
    cvt(gw, gwb, nGW);
    cvt(uw, uwb, nGW);
    cvt(dw, dwb, nDW);

    // 3) shared expert gate/up fused (+SiLU): hs = silu(x@Wg^T)*(x@Wu^T)
    //    Wg/Wu are [HS, C] = [N, K] row-major  -> BT=true
    gateup_kernel<true><<<dim3(HS_DIM / 128, S_TOK / 64, 1), 256, 0, stream>>>(
        xbf, C_DIM, sgwb, suwb, C_DIM, 0, hsbf, HS_DIM, 0, nullptr, C_DIM);

    // 4) routed gate/up fused (+SiLU, +combine weight): h[e] per expert
    //    gate_w/up_w are [E, C, H] = [K, N] per expert -> BT=false
    gateup_kernel<false><<<dim3(H_DIM / 128, S_TOK / 64, E_NUM), 256, 0, stream>>>(
        xbf, C_DIM, gwb, uwb, H_DIM, (long)C_DIM * H_DIM,
        hbf, H_DIM, (long)S_TOK * H_DIM, combine, C_DIM);

    // 5) shared down: out = hs @ Wd^T, Wd [C, HS] = [N, K] -> BT=true, beta=0
    down_kernel<true, false><<<dim3(C_DIM / 128, S_TOK / 64), 256, 0, stream>>>(
        hsbf, HS_DIM, 0, sdwb, HS_DIM, 0, out, C_DIM, HS_DIM, 1);

    // 6) routed down: out += sum_e h[e] @ down_w[e], down_w [H, C] = [K, N] -> BT=false
    down_kernel<false, true><<<dim3(C_DIM / 128, S_TOK / 64), 256, 0, stream>>>(
        hbf, H_DIM, (long)S_TOK * H_DIM, dwb, C_DIM, (long)H_DIM * C_DIM,
        out, C_DIM, H_DIM, E_NUM);

    (void)in_sizes; (void)n_in; (void)out_size; (void)ws_size;
}
